// ChebyshevFeatures_66700842106972
// MI455X (gfx1250) — compile-verified, matrix operand order checked by reading, not run on hardware
//
#include <hip/hip_runtime.h>

typedef int v8i __attribute__((ext_vector_type(8)));

#define NIMG   1600      // B*P = 16*100 images per scale
#define NPAT   100       // patches per sample
#define NB     16        // batch
#define MAXS2  1681      // 41*41

// ---------------------------------------------------------------------------
// Kernel 1: binarize against center pixel (faithful TF reshape semantics).
// binm is stored per-image with pitch SP (multiple of 64), zero padded so the
// WMMA count reduction can run uniform 64-byte K-chunks.
// ---------------------------------------------------------------------------
__global__ void binm_kernel(const float* __restrict__ x,
                            unsigned char* __restrict__ binm,
                            int s, int SP, int total) {
    int tid = blockIdx.x * blockDim.x + threadIdx.x;
    if (tid >= total) return;
    int g   = tid / SP;          // image index b*100+p
    int off = tid - g * SP;      // position inside padded patch
    unsigned char v = 0;
    int s2 = s * s;
    if (off < s2) {
        int r1 = off / s;
        int c2 = off - r1 * s;
        int q  = g * s + r1;                 // flat row over (b*P*s) rows
        int i0 = q / NPAT;                   // reshape(b*s, p, s, c) row group
        long crow  = (long)i0 * NPAT + 49;   // center patch index (p-1)//2
        long pbase = ((long)q * s + c2) * 3;
        long cbase = (crow * s + (s - 1) / 2) * 3;
        float d0 = fabsf(x[pbase + 0] - x[cbase + 0]);
        float d1 = fabsf(x[pbase + 1] - x[cbase + 1]);
        float d2 = fabsf(x[pbase + 2] - x[cbase + 2]);
        float mx = fmaxf(d0, fmaxf(d1, d2));
        v = (mx <= (float)s) ? 1 : 0;
    }
    binm[tid] = v;
}

// ---------------------------------------------------------------------------
// Kernel 2: per-patch foreground counts via V_WMMA_I32_16X16X64_IU8.
// counts = binm(16 rows x K) * ones(K x 16): every D column = row sum.
// B = all-ones makes the A K-layout permutation-irrelevant, so lane L (<16)
// loads bytes [0,32) and lane L+16 loads bytes [32,64) of row L per chunk.
// One wave per 16 patches; EXEC is all-1s through the WMMA loop.
// ---------------------------------------------------------------------------
__global__ void __launch_bounds__(32)
counts_wmma_kernel(const unsigned char* __restrict__ binm,
                   int* __restrict__ counts, int SP) {
    int blk  = blockIdx.x;          // 100 blocks x 16 patches
    int lane = threadIdx.x;         // wave32
    int m    = lane & 15;
    int half = lane >> 4;
    const unsigned char* rowp =
        binm + (size_t)(blk * 16 + m) * SP + half * 32;

    v8i acc  = {0, 0, 0, 0, 0, 0, 0, 0};
    const int o = 0x01010101;
    v8i ones = {o, o, o, o, o, o, o, o};

    int chunks = SP >> 6;
    for (int c = 0; c < chunks; ++c) {
        v8i a = *(const v8i*)(rowp + (size_t)c * 64);
        acc = __builtin_amdgcn_wmma_i32_16x16x64_iu8(
            false, a, false, ones, acc, false, false);
    }
    // D layout: VGPR j = row j (lanes 0-15, N=lane) / row j+8 (lanes 16-31).
    if (lane == 0) {
        #pragma unroll
        for (int j = 0; j < 8; ++j) counts[blk * 16 + j] = acc[j];
    } else if (lane == 16) {
        #pragma unroll
        for (int j = 0; j < 8; ++j) counts[blk * 16 + 8 + j] = acc[j];
    }
}

// ---------------------------------------------------------------------------
// Kernel 3: fd / lac / pq from counts. hist collapses analytically:
//   fd  = mean_p 1/max(c,1)
//   lac = (E[n^2]-E[n]^2)/E[n]^2  with n = max(c,1)
//   pq  = mean_p [c/(s*s) >= 0.59275]
// One wave per batch sample.
// ---------------------------------------------------------------------------
__global__ void __launch_bounds__(32)
stats_kernel(const int* __restrict__ counts, float* __restrict__ out,
             int s, int si) {
    int b = blockIdx.x;
    int lane = threadIdx.x;
    float s2f = (float)(s * s);
    float sinv = 0.f, sn = 0.f, sn2 = 0.f, sq = 0.f;
    for (int p = lane; p < NPAT; p += 32) {
        int c = counts[b * NPAT + p];
        float n = (float)(c > 0 ? c : 1);
        sinv += 1.0f / n;
        sn   += n;
        sn2  += n * n;
        sq   += (((float)c / s2f) >= 0.59275f) ? 1.0f : 0.0f;
    }
    #pragma unroll
    for (int o = 16; o > 0; o >>= 1) {
        sinv += __shfl_xor(sinv, o, 32);
        sn   += __shfl_xor(sn,   o, 32);
        sn2  += __shfl_xor(sn2,  o, 32);
        sq   += __shfl_xor(sq,   o, 32);
    }
    if (lane == 0) {
        float fd  = sinv * 0.01f;
        float Em  = sn  * 0.01f;
        float E2  = sn2 * 0.01f;
        float mu1 = Em * Em;
        float lac = (E2 - mu1) / mu1;
        out[b * 100 + 20 + si] = sq * 0.01f;   // pq
        out[b * 100 + 60 + si] = lac;          // lac
        out[b * 100 + 80 + si] = fd;           // fd
    }
}

// ---------------------------------------------------------------------------
// Kernel 4: 4-connected components per image in LDS (min-label propagation,
// chaotic Gauss-Seidel sweeps -> same unique fixed point as reference Jacobi).
// Emits per image: #components, background pixels, largest component size.
// ---------------------------------------------------------------------------
__global__ void __launch_bounds__(256)
cc_kernel(const unsigned char* __restrict__ binm,
          int* __restrict__ ncomp, int* __restrict__ bgpix,
          int* __restrict__ maxcomp, int s, int SP) {
    __shared__ int lab[MAXS2];
    __shared__ int hist[MAXS2];
    __shared__ int changed, sh_ncomp, sh_fg, sh_max;

    int g  = blockIdx.x;
    int t  = threadIdx.x;
    int s2 = s * s;
    int BIG = s2 + 1;

    for (int j = t; j < s2; j += 256) {
        lab[j]  = binm[(size_t)g * SP + j] ? (j + 1) : BIG;
        hist[j] = 0;
    }
    if (t == 0) { sh_ncomp = 0; sh_fg = 0; sh_max = 0; }
    __syncthreads();

    while (true) {
        if (t == 0) changed = 0;
        __syncthreads();
        for (int j = t; j < s2; j += 256) {
            int l = lab[j];
            if (l != BIG) {
                int r = j / s, c = j - r * s;
                int m = l;
                if (r > 0)     m = min(m, lab[j - s]);
                if (r < s - 1) m = min(m, lab[j + s]);
                if (c > 0)     m = min(m, lab[j - 1]);
                if (c < s - 1) m = min(m, lab[j + 1]);
                if (m < l) { lab[j] = m; changed = 1; }
            }
        }
        __syncthreads();
        int ch = changed;
        __syncthreads();          // protect reset vs. this read
        if (!ch) break;
    }

    int lcomp = 0, lfg = 0;
    for (int j = t; j < s2; j += 256) {
        int l = lab[j];
        if (l != BIG) {
            lfg++;
            if (l == j + 1) lcomp++;       // component representative
            atomicAdd(&hist[l - 1], 1);    // component size histogram
        }
    }
    atomicAdd(&sh_ncomp, lcomp);
    atomicAdd(&sh_fg, lfg);
    __syncthreads();
    int lmax = 0;
    for (int j = t; j < s2; j += 256) lmax = max(lmax, hist[j]);
    atomicMax(&sh_max, lmax);
    __syncthreads();
    if (t == 0) {
        ncomp[g]   = sh_ncomp;
        bgpix[g]   = s2 - sh_fg;
        maxcomp[g] = sh_max;
    }
}

// ---------------------------------------------------------------------------
// Kernel 5: global (batch-wide) cumsum of component counts (tfa consecutive
// labels), then pc = mean_p maxlab, pm = max(total bg, largest component).
// ---------------------------------------------------------------------------
__global__ void __launch_bounds__(256)
cc_final_kernel(const int* __restrict__ ncomp, const int* __restrict__ bgpix,
                const int* __restrict__ maxcomp, float* __restrict__ out,
                int si) {
    __shared__ int pref[NIMG];
    int t = threadIdx.x;
    if (t == 0) {
        int run = 0;
        for (int i = 0; i < NIMG; ++i) { run += ncomp[i]; pref[i] = run; }
    }
    __syncthreads();
    if (t < NB) {
        int b = t;
        long suml = 0, bgs = 0;
        int mc = 0;
        for (int p = 0; p < NPAT; ++p) {
            int i = b * NPAT + p;
            if (ncomp[i] > 0) suml += pref[i];
            bgs += bgpix[i];
            mc = max(mc, maxcomp[i]);
        }
        out[b * 100 + 0  + si] = (float)suml / 100.0f;           // pc
        out[b * 100 + 40 + si] = fmaxf((float)bgs, (float)mc);   // pm
    }
}

// ---------------------------------------------------------------------------
// Host side: 20 scales processed sequentially on one stream -> single reusable
// workspace region (max 1600*1728 B binm + 4*1600*4 B scalars < 2.8 MB).
// ---------------------------------------------------------------------------
extern "C" void kernel_launch(void* const* d_in, const int* in_sizes, int n_in,
                              void* d_out, int out_size, void* d_ws, size_t ws_size,
                              hipStream_t stream) {
    (void)in_sizes; (void)n_in; (void)out_size; (void)ws_size;
    float* out = (float*)d_out;

    unsigned char* binm = (unsigned char*)d_ws;
    const size_t BINM_BYTES = (size_t)NIMG * 1728;  // max padded, 64B aligned
    int* counts = (int*)((char*)d_ws + BINM_BYTES);
    int* ncomp  = counts + NIMG;
    int* bgw    = ncomp  + NIMG;
    int* maxc   = bgw    + NIMG;

    for (int i = 0; i < 20; ++i) {
        int s  = 3 + 2 * i;
        int s2 = s * s;
        int SP = ((s2 + 63) / 64) * 64;
        const float* x = (const float*)d_in[i];
        int total = NIMG * SP;

        binm_kernel<<<(total + 255) / 256, 256, 0, stream>>>(x, binm, s, SP, total);
        counts_wmma_kernel<<<NIMG / 16, 32, 0, stream>>>(binm, counts, SP);
        stats_kernel<<<NB, 32, 0, stream>>>(counts, out, s, i);
        cc_kernel<<<NIMG, 256, 0, stream>>>(binm, ncomp, bgw, maxc, s, SP);
        cc_final_kernel<<<1, 256, 0, stream>>>(ncomp, bgw, maxc, out, i);
    }
}
